// TransformerEncoderLayer_32014686225065
// MI455X (gfx1250) — matrix operand order repaired and checked by reading, not compiled, not run on hardware
//
#include <hip/hip_runtime.h>

// ---------------------------------------------------------------------------
// Transformer encoder layer for MI455X (gfx1250):
//  - all GEMMs on v_wmma_f32_16x16x32_bf16 (wave32 WMMA, fp32 accumulate)
//  - flash attention: online softmax, row-max via DPP, row-sum via ones-WMMA
//  - async global->LDS staging (ASYNCcnt) for straight-copy tiles
//  - global_prefetch on next tiles
// ---------------------------------------------------------------------------

#define BB     2
#define SS     2048
#define DMODEL 1024
#define NH     16
#define HDm    64
#define DFFm   4096

typedef unsigned short u16;
typedef __bf16 v8bf  __attribute__((ext_vector_type(8)));
typedef __bf16 v16bf __attribute__((ext_vector_type(16)));
typedef float  v8f   __attribute__((ext_vector_type(8)));

#define WMMA_BF16(a, b, c) \
  __builtin_amdgcn_wmma_f32_16x16x32_bf16(false, (a), false, (b), (short)0, (c), false, false)

__device__ __forceinline__ u16 f2bf(float f) {
  unsigned u = __float_as_uint(f);
  unsigned r = 0x7fffu + ((u >> 16) & 1u);   // round-to-nearest-even
  return (u16)((u + r) >> 16);
}

__device__ __forceinline__ v8f vzero() {
  v8f z;
#pragma unroll
  for (int j = 0; j < 8; ++j) z[j] = 0.0f;
  return z;
}

__device__ __forceinline__ v16bf frag16(const u16* p0, const u16* p1) {
  v8bf lo = *(const v8bf*)p0;
  v8bf hi = *(const v8bf*)p1;
  v16bf f;
#pragma unroll
  for (int i = 0; i < 8; ++i) { f[i] = lo[i]; f[i + 8] = hi[i]; }
  return f;
}

// A fragment (16x32 MxK, 16-bit; ISA 7.12.2): lane holds row M=lane&15,
// elems 0..7 -> K = lh*8+i, elems 8..15 -> K = 16 + lh*8 + (i-8), lh = lane>>4.
__device__ __forceinline__ v16bf a_frag(const u16* row, int lh) {
  return frag16(row + lh * 8, row + 16 + lh * 8);
}
// B fragment (32x16 KxN): lane holds col N=lane&15, same K order as A fragment.
__device__ __forceinline__ v16bf b_frag(const u16* col, int lh) {
  return frag16(col + lh * 8, col + 16 + lh * 8);
}

__device__ __forceinline__ v16bf ones_frag() {
  union { u16 u; __bf16 b; } c;
  c.u = 0x3F80;   // bf16 1.0
  v16bf f;
#pragma unroll
  for (int i = 0; i < 16; ++i) f[i] = c.b;
  return f;
}

// Async global->LDS copy, 16B per lane (GLOBAL_LOAD_ASYNC_TO_LDS_B128, GVS):
// lds = LDS byte address (low 32 bits of generic pointer), goff = per-lane
// byte offset, base = uniform 64-bit global base in SGPRs. Tracked by ASYNCcnt.
__device__ __forceinline__ void async_load_b128(unsigned lds, unsigned goff,
                                                unsigned long long base) {
  asm volatile("global_load_async_to_lds_b128 %0, %1, %2"
               :: "v"(lds), "v"(goff), "s"(base) : "memory");
}
__device__ __forceinline__ void wait_async0() {
  asm volatile("s_wait_asynccnt 0" ::: "memory");
}

// max-reduce across 16 lanes with DPP (no LDS traffic):
// quad_perm xor1, quad_perm xor2, row_half_mirror, row_mirror
template <int CTRL>
__device__ __forceinline__ float fmax_dpp(float v) {
  int x = __builtin_amdgcn_update_dpp(0, __float_as_int(v), CTRL, 0xf, 0xf, true);
  return fmaxf(v, __int_as_float(x));
}
__device__ __forceinline__ float rowmax16(float v) {
  v = fmax_dpp<0xB1>(v);    // quad_perm(1,0,3,2)  == xor 1
  v = fmax_dpp<0x4E>(v);    // quad_perm(2,3,0,1)  == xor 2
  v = fmax_dpp<0x141>(v);   // row_half_mirror     (reduce 8)
  v = fmax_dpp<0x140>(v);   // row_mirror          (reduce 16)
  return v;
}

// --------------------------- conversion kernels ----------------------------

__global__ void cvt_bf16_kernel(const float* __restrict__ src,
                                u16* __restrict__ dst, int n) {
  int i = blockIdx.x * blockDim.x + threadIdx.x;
  if (i < n) dst[i] = f2bf(src[i]);
}

// [H, D, E] fp32 -> [D, H*E] bf16
__global__ void repack_hw_kernel(const float* __restrict__ w,
                                 u16* __restrict__ dst, int H_, int D_, int E_) {
  int i = blockIdx.x * blockDim.x + threadIdx.x;
  int n = H_ * D_ * E_;
  if (i >= n) return;
  int h = i / (D_ * E_);
  int rem = i - h * D_ * E_;
  int d = rem / E_;
  int e = rem - d * E_;
  dst[(size_t)d * (H_ * E_) + h * E_ + e] = f2bf(w[i]);
}

// ------------------------------ WMMA GEMM ----------------------------------
// C[M,N] = A[M,K]*B[K,N] + bias;  A,B bf16 row-major, fp32 accumulate.
// WG tile 128x128, K-step 64; 8 waves as 4(M) x 2(N); wave tile 32x64
// -> 16 WMMA per wave per K-step against 12 LDS fragment loads.
// A tile staged via async global->LDS; B tile transposed through registers.

template <bool RELU, bool BF16OUT>
__global__ __launch_bounds__(256) void gemm_bf16_kernel(
    const u16* __restrict__ A, const u16* __restrict__ Bw,
    const float* __restrict__ bias,
    float* __restrict__ Cf, u16* __restrict__ Cb,
    int M, int N, int K) {
  __shared__ u16 As[128][64];    // row-major A tile        (16 KB)
  __shared__ u16 Bs[128][64];    // transposed B: Bs[n][k]  (16 KB)

  const int lane = threadIdx.x & 31;
  const int wave = threadIdx.x >> 5;
  const int lh = lane >> 4;
  const int li = lane & 15;
  const int wm = wave >> 1;      // 0..3
  const int wn = wave & 1;       // 0..1
  const int m0 = blockIdx.x * 128;
  const int n0 = blockIdx.y * 128;
  const unsigned long long abase = (unsigned long long)(uintptr_t)A;

  v8f acc[8];
#pragma unroll
  for (int i = 0; i < 8; ++i) acc[i] = vzero();

  for (int k0 = 0; k0 < K; k0 += 64) {
    __syncthreads();
    // stage A: 128x64 elems, 4 async 16B copies per thread (no VGPR round-trip)
#pragma unroll
    for (int c = 0; c < 4; ++c) {
      int idx = (threadIdx.x + c * 256) * 8;
      int am = idx >> 6, ak = idx & 63;
      async_load_b128((unsigned)(uintptr_t)&As[am][ak],
                      (unsigned)(((size_t)(m0 + am) * K + k0 + ak) * 2), abase);
    }
    // stage B transposed: each thread moves a 4k x 8n block
    {
      int kb = (threadIdx.x & 15) * 4;
      int nb = (threadIdx.x >> 4) * 8;
      union { uint4 v; u16 h[8]; } g[4];
#pragma unroll
      for (int j = 0; j < 4; ++j)
        g[j].v = *(const uint4*)(Bw + (size_t)(k0 + kb + j) * N + n0 + nb);
#pragma unroll
      for (int i = 0; i < 8; ++i) {
        union { uint2 v; u16 h[4]; } t;
#pragma unroll
        for (int j = 0; j < 4; ++j) t.h[j] = g[j].h[i];
        *(uint2*)&Bs[nb + i][kb] = t.v;
      }
    }
    // prefetch next K slab into L2 (global_prefetch_b8)
    if (k0 + 64 < K) {
      __builtin_prefetch(A + (size_t)(m0 + (threadIdx.x >> 1)) * K + k0 + 64, 0, 1);
      __builtin_prefetch(Bw + (size_t)(k0 + 64 + (threadIdx.x & 63)) * N + n0 +
                             (threadIdx.x >> 6) * 32, 0, 1);
    }
    wait_async0();       // own async LDS writes landed
    __syncthreads();     // => all waves' writes landed

#pragma unroll
    for (int kc = 0; kc < 2; ++kc) {
      v16bf a0 = a_frag(&As[wm * 32 + li][kc * 32], lh);
      v16bf a1 = a_frag(&As[wm * 32 + 16 + li][kc * 32], lh);
#pragma unroll
      for (int nt = 0; nt < 4; ++nt) {
        v16bf bf = b_frag(&Bs[wn * 64 + nt * 16 + li][kc * 32], lh);
        acc[nt]     = WMMA_BF16(a0, bf, acc[nt]);
        acc[4 + nt] = WMMA_BF16(a1, bf, acc[4 + nt]);
      }
    }
  }

  // epilogue: C layout (ISA 7.12.2): VGPR r -> M = r + 8*lh, N = lane&15
#pragma unroll
  for (int mi = 0; mi < 2; ++mi) {
#pragma unroll
    for (int nt = 0; nt < 4; ++nt) {
      int n = n0 + wn * 64 + nt * 16 + li;
      float bv = bias[n];
#pragma unroll
      for (int r = 0; r < 8; ++r) {
        int m = m0 + wm * 32 + mi * 16 + r + 8 * lh;
        float v = acc[mi * 4 + nt][r] + bv;
        if (RELU) v = fmaxf(v, 0.0f);
        if (BF16OUT) Cb[(size_t)m * N + n] = f2bf(v);
        else         Cf[(size_t)m * N + n] = v;
      }
    }
  }
}

// --------------------------- flash attention -------------------------------
// Q,K,V,O bf16 [B, S, H*HD]. One WG = 128 q rows of one (b,h); each wave owns
// 16 q rows. Per 32 kv rows: 4 WMMA (QK^T), online softmax (DPP row-max,
// ones-WMMA row-sum), 4 WMMA (P.V). K tile staged async; V transposed.

__global__ __launch_bounds__(256) void flash_attn_kernel(
    const u16* __restrict__ Qg, const u16* __restrict__ Kg,
    const u16* __restrict__ Vg, u16* __restrict__ Og) {
  __shared__ u16 Ksh[32][64];        // [t][e]
  __shared__ u16 Vsh[64][32];        // transposed: [e][t]
  __shared__ u16 Psh[8][16][32];     // per-wave P transpose scratch

  const int lane = threadIdx.x & 31;
  const int wave = threadIdx.x >> 5;
  const int lh = lane >> 4;
  const int li = lane & 15;

  const int qtiles = SS / 128;
  int bid = blockIdx.x;
  int qt = bid % qtiles;
  int h = (bid / qtiles) % NH;
  int b = bid / (qtiles * NH);

  const size_t rs = DMODEL;
  const u16* Qp = Qg + (size_t)b * SS * rs + (size_t)h * HDm;
  const u16* Kp = Kg + (size_t)b * SS * rs + (size_t)h * HDm;
  const u16* Vp = Vg + (size_t)b * SS * rs + (size_t)h * HDm;
  const unsigned long long kbase = (unsigned long long)(uintptr_t)Kp;
  const int q0 = qt * 128 + wave * 16;

  v16bf qf[2];
#pragma unroll
  for (int kc = 0; kc < 2; ++kc)
    qf[kc] = a_frag(Qp + (size_t)(q0 + li) * rs + kc * 32, lh);

  const v16bf onesf = ones_frag();

  v8f acc[4];
#pragma unroll
  for (int i = 0; i < 4; ++i) acc[i] = vzero();
  v8f accl = vzero();                 // softmax denominator via ones-WMMA
  float mrow[8];
#pragma unroll
  for (int r = 0; r < 8; ++r) mrow[r] = -3.0e38f;

  const float scale = 0.125f;   // 1/sqrt(64)

  for (int t0 = 0; t0 < SS; t0 += 32) {
    __syncthreads();
    {   // stage 32x64 K rows (async) + transposed V (register transpose)
      int idx = threadIdx.x * 8;
      int tr = idx >> 6, te = idx & 63;
      async_load_b128((unsigned)(uintptr_t)&Ksh[tr][te],
                      (unsigned)(((size_t)(t0 + tr) * rs + te) * 2), kbase);
      union { uint4 v; u16 h[8]; } tmp;
      tmp.v = *(const uint4*)(Vp + (size_t)(t0 + tr) * rs + te);
#pragma unroll
      for (int i = 0; i < 8; ++i) Vsh[te + i][tr] = tmp.h[i];
      if (t0 + 32 < SS) {   // prefetch next KV slab
        __builtin_prefetch(Kp + (size_t)(t0 + 32 + tr) * rs + te, 0, 1);
        __builtin_prefetch(Vp + (size_t)(t0 + 32 + tr) * rs + te, 0, 1);
      }
    }
    wait_async0();
    __syncthreads();

    // scores: S[16 q, 32 t] = Q(16x64) . K^T(64x32); B elem (e,t) = K[t][e]
    v8f sc0 = vzero(), sc1 = vzero();
#pragma unroll
    for (int kc = 0; kc < 2; ++kc) {
      v16bf bf0 = b_frag(&Ksh[li][kc * 32], lh);
      v16bf bf1 = b_frag(&Ksh[16 + li][kc * 32], lh);
      sc0 = WMMA_BF16(qf[kc], bf0, sc0);
      sc1 = WMMA_BF16(qf[kc], bf1, sc1);
    }

    // online softmax; row m = r + 8*lh lives across 16 lanes of one half
#pragma unroll
    for (int r = 0; r < 8; ++r) {
      float s0 = sc0[r] * scale, s1 = sc1[r] * scale;
      float mx = rowmax16(fmaxf(s0, s1));
      float mnew = fmaxf(mrow[r], mx);
      float alpha = __expf(mrow[r] - mnew);
      float p0 = __expf(s0 - mnew);
      float p1 = __expf(s1 - mnew);
      int mr = r + 8 * lh;
      Psh[wave][mr][li] = f2bf(p0);
      Psh[wave][mr][16 + li] = f2bf(p1);
      mrow[r] = mnew;
      accl[r] *= alpha;
#pragma unroll
      for (int eg = 0; eg < 4; ++eg) acc[eg][r] *= alpha;
    }

    // wave-private scratch: DS ops are in-order per wave; make waits explicit
    asm volatile("s_wait_dscnt 0" ::: "memory");

    v16bf pf = a_frag(&Psh[wave][li][0], lh);
    accl = WMMA_BF16(pf, onesf, accl);   // row sums of P (every lane/col)
#pragma unroll
    for (int eg = 0; eg < 4; ++eg) {
      v16bf vf = b_frag(&Vsh[eg * 16 + li][0], lh);
      acc[eg] = WMMA_BF16(pf, vf, acc[eg]);
    }
  }

#pragma unroll
  for (int eg = 0; eg < 4; ++eg) {
#pragma unroll
    for (int r = 0; r < 8; ++r) {
      int s = q0 + r + 8 * lh;
      float o = acc[eg][r] / accl[r];
      Og[(size_t)(b * SS + s) * rs + h * HDm + eg * 16 + li] = f2bf(o);
    }
  }
}

// --------------------------- add + layernorm -------------------------------

__global__ __launch_bounds__(256) void add_ln_kernel(
    const float* __restrict__ X, const float* __restrict__ R,
    const float* __restrict__ g, const float* __restrict__ bta,
    float* __restrict__ Yf, u16* __restrict__ Yb) {
  __shared__ float rsum[8], rsq[8], stat[2];
  const int row = blockIdx.x;
  const float* xr = X + (size_t)row * DMODEL;
  const float* rr = R + (size_t)row * DMODEL;

  float s1 = 0.0f, s2 = 0.0f;
  for (int i = threadIdx.x; i < DMODEL; i += 256) {
    float v = xr[i] + rr[i];
    s1 += v; s2 += v * v;
  }
#pragma unroll
  for (int m = 16; m; m >>= 1) {
    s1 += __shfl_xor(s1, m, 32);
    s2 += __shfl_xor(s2, m, 32);
  }
  if ((threadIdx.x & 31) == 0) {
    rsum[threadIdx.x >> 5] = s1;
    rsq[threadIdx.x >> 5] = s2;
  }
  __syncthreads();
  if (threadIdx.x == 0) {
    float a = 0.0f, q = 0.0f;
    for (int i = 0; i < 8; ++i) { a += rsum[i]; q += rsq[i]; }
    float mean = a / DMODEL;
    float var = q / DMODEL - mean * mean;
    stat[0] = mean;
    stat[1] = rsqrtf(var + 1e-5f);
  }
  __syncthreads();
  float mean = stat[0], rstd = stat[1];
  for (int i = threadIdx.x; i < DMODEL; i += 256) {
    float v = (xr[i] + rr[i] - mean) * rstd * g[i] + bta[i];
    Yf[(size_t)row * DMODEL + i] = v;
    if (Yb) Yb[(size_t)row * DMODEL + i] = f2bf(v);
  }
}

// ------------------------------- launcher ----------------------------------

extern "C" void kernel_launch(void* const* d_in, const int* in_sizes, int n_in,
                              void* d_out, int out_size, void* d_ws,
                              size_t ws_size, hipStream_t stream) {
  (void)in_sizes; (void)n_in; (void)out_size; (void)ws_size;

  const float* X   = (const float*)d_in[0];
  const float* Wq  = (const float*)d_in[1];
  const float* bq  = (const float*)d_in[2];
  const float* Wk  = (const float*)d_in[3];
  const float* bk  = (const float*)d_in[4];
  const float* Wv  = (const float*)d_in[5];
  const float* bv  = (const float*)d_in[6];
  const float* Wo  = (const float*)d_in[7];
  const float* bo  = (const float*)d_in[8];
  const float* g1  = (const float*)d_in[9];
  const float* be1 = (const float*)d_in[10];
  const float* W1  = (const float*)d_in[11];
  const float* b1  = (const float*)d_in[12];
  const float* W2  = (const float*)d_in[13];
  const float* b2  = (const float*)d_in[14];
  const float* g2  = (const float*)d_in[15];
  const float* be2 = (const float*)d_in[16];
  float* out = (float*)d_out;

  const int M = BB * SS;   // 4096 rows

  char* p = (char*)d_ws;
  size_t off = 0;
  auto take = [&](size_t bytes) -> void* {
    void* q = p + off;
    off = (off + bytes + 255) & ~(size_t)255;
    return q;
  };
  u16* Xb   = (u16*)take((size_t)M * DMODEL * 2);
  u16* Wqb  = (u16*)take((size_t)DMODEL * DMODEL * 2);
  u16* Wkb  = (u16*)take((size_t)DMODEL * DMODEL * 2);
  u16* Wvb  = (u16*)take((size_t)DMODEL * DMODEL * 2);
  u16* Wob  = (u16*)take((size_t)DMODEL * DMODEL * 2);
  u16* W1b  = (u16*)take((size_t)DMODEL * DFFm * 2);
  u16* W2b  = (u16*)take((size_t)DFFm * DMODEL * 2);
  u16* Qb   = (u16*)take((size_t)M * DMODEL * 2);
  u16* Kb   = (u16*)take((size_t)M * DMODEL * 2);
  u16* Vb   = (u16*)take((size_t)M * DMODEL * 2);
  u16* Ob   = (u16*)take((size_t)M * DMODEL * 2);
  float* attn = (float*)take((size_t)M * DMODEL * 4);   // reused for FFN2 out
  float* yf   = (float*)take((size_t)M * DMODEL * 4);
  u16* yb     = (u16*)take((size_t)M * DMODEL * 2);
  u16* hid    = (u16*)take((size_t)M * DFFm * 2);

  dim3 blk(256);
  int n;

  n = M * DMODEL;
  cvt_bf16_kernel<<<(n + 255) / 256, blk, 0, stream>>>(X, Xb, n);
  n = DMODEL * DMODEL;
  cvt_bf16_kernel<<<(n + 255) / 256, blk, 0, stream>>>(Wo, Wob, n);
  n = DMODEL * DFFm;
  cvt_bf16_kernel<<<(n + 255) / 256, blk, 0, stream>>>(W1, W1b, n);
  cvt_bf16_kernel<<<(n + 255) / 256, blk, 0, stream>>>(W2, W2b, n);
  n = NH * DMODEL * HDm;
  repack_hw_kernel<<<(n + 255) / 256, blk, 0, stream>>>(Wq, Wqb, NH, DMODEL, HDm);
  repack_hw_kernel<<<(n + 255) / 256, blk, 0, stream>>>(Wk, Wkb, NH, DMODEL, HDm);
  repack_hw_kernel<<<(n + 255) / 256, blk, 0, stream>>>(Wv, Wvb, NH, DMODEL, HDm);

  dim3 gridP(M / 128, DMODEL / 128);
  gemm_bf16_kernel<false, true><<<gridP, blk, 0, stream>>>(
      Xb, Wqb, bq, nullptr, Qb, M, DMODEL, DMODEL);
  gemm_bf16_kernel<false, true><<<gridP, blk, 0, stream>>>(
      Xb, Wkb, bk, nullptr, Kb, M, DMODEL, DMODEL);
  gemm_bf16_kernel<false, true><<<gridP, blk, 0, stream>>>(
      Xb, Wvb, bv, nullptr, Vb, M, DMODEL, DMODEL);

  flash_attn_kernel<<<BB * NH * (SS / 128), blk, 0, stream>>>(Qb, Kb, Vb, Ob);

  gemm_bf16_kernel<false, false><<<gridP, blk, 0, stream>>>(
      Ob, Wob, bo, attn, nullptr, M, DMODEL, DMODEL);

  add_ln_kernel<<<M, blk, 0, stream>>>(X, attn, g1, be1, yf, yb);

  dim3 gridF(M / 128, DFFm / 128);
  gemm_bf16_kernel<true, true><<<gridF, blk, 0, stream>>>(
      yb, W1b, b1, nullptr, hid, M, DFFm, DMODEL);
  gemm_bf16_kernel<false, false><<<gridP, blk, 0, stream>>>(
      hid, W2b, b2, attn, nullptr, M, DMODEL, DFFm);

  add_ln_kernel<<<M, blk, 0, stream>>>(yf, attn, g2, be2, out, nullptr);
}
